// DCNv3Module_82188494176856
// MI455X (gfx1250) — matrix operand order repaired and checked by reading, not compiled, not run on hardware
//
#include <hip/hip_runtime.h>
#include <hip/hip_bf16.h>
#include <hip/hip_fp16.h>

typedef __attribute__((ext_vector_type(16))) _Float16     v16h;
typedef __attribute__((ext_vector_type(8)))  float        v8f;
typedef __attribute__((ext_vector_type(4)))  unsigned int v4u;
typedef __attribute__((ext_vector_type(8)))  int          v8i;
typedef __attribute__((ext_vector_type(4)))  int          v4i;

namespace {
constexpr int KS = 3, PAD = 1, G = 4, C_IN = 64, C_OUT = 64, K = 9;
constexpr int Bn = 4, H = 128, W = 128, HW = H * W, NPIX = Bn * HW; // 65536
constexpr int OMC = 108;      // G*3*K offset/mask channels
constexpr int OMS = 112;      // padded to 7 tiles of 16
constexpr int CG  = 16;       // channels per group (in == out)
constexpr float EPS = 1e-5f;
constexpr int PWOM_N  = 7 * 2 * 32 * 16;  // 7168 f16
constexpr int PWDCN_N = 4 * 5 * 32 * 16;  // 10240 f16
constexpr int XROW = 130;                 // LDS row stride (128 + 2 TDM pad dw)
}

// ---------------------------------------------------------------------------
// Kernel 0: pre-pack weights into WMMA B-fragment order (f16, zero padded)
// ---------------------------------------------------------------------------
__global__ __launch_bounds__(256) void prepack(
    const float* __restrict__ w_om, const float* __restrict__ b_om,
    const float* __restrict__ dcn_w,
    _Float16* __restrict__ pwom, _Float16* __restrict__ pwdcn,
    float* __restrict__ b112)
{
    for (int t = threadIdx.x; t < OMS; t += 256)
        b112[t] = (t < OMC) ? b_om[t] : 0.f;

    for (int idx = threadIdx.x; idx < PWOM_N; idx += 256) {
        int e    = idx & 15;
        int lane = (idx >> 4) & 31;
        int ch   = (idx >> 9) & 1;
        int nt   = idx >> 10;
        int o    = nt * 16 + (lane & 15);
        int kk   = ch * 32 + (e >> 3) * 16 + (lane >> 4) * 8 + (e & 7);
        pwom[idx] = (o < OMC) ? (_Float16)w_om[(size_t)o * C_IN + kk]
                              : (_Float16)0.f;
    }

    for (int idx = threadIdx.x; idx < PWDCN_N; idx += 256) {
        int e    = idx & 15;
        int lane = (idx >> 4) & 31;
        int rem  = idx >> 9;          // g*5 + chunk
        int c    = rem % 5;
        int g    = rem / 5;
        int co   = lane & 15;
        int r    = c * 32 + (e >> 3) * 16 + (lane >> 4) * 8 + (e & 7);
        pwdcn[idx] = (r < CG * K)
            ? (_Float16)dcn_w[(size_t)((g * CG + co) * CG + (r & 15)) * 9 + (r >> 4)]
            : (_Float16)0.f;
    }
}

// ---------------------------------------------------------------------------
// Kernel 1: om = x * w_om^T + b_om.
// One block = 128 consecutive pixels. The [64ch x 128px] f32 x-tile is staged
// into LDS by ONE Tensor-Data-Mover descriptor (2D tile, dim0=128 contiguous
// pixels, dim1=64 channel rows of stride HW, with 2-dword LDS padding per
// 128-dword row so the two lane halves hit disjoint banks). Each wave then
// builds its A fragments once and sweeps all 7 output n-tiles (14 WMMAs).
// ---------------------------------------------------------------------------
__global__ __launch_bounds__(256) void om_gemm_wmma(
    const float* __restrict__ x, const _Float16* __restrict__ pwom,
    const float* __restrict__ b112, float* __restrict__ om)
{
    __shared__ float shx[C_IN * XROW];    // 64 * 130 dwords = 33,280 B
    const int lane = threadIdx.x & 31;
    const int wv   = threadIdx.x >> 5;
    const int m    = lane & 15;
    const int hl   = lane >> 4;
    const int pix0 = blockIdx.x * 128;
    const int b    = pix0 >> 14;
    const int hw0  = pix0 & (HW - 1);

    if (wv == 0) {
        unsigned long long ga =
            (unsigned long long)(uintptr_t)(x + (size_t)b * C_IN * HW + hw0);
        unsigned lds = (unsigned)(uintptr_t)(&shx[0]);
        // D# group 0: count=1 | lds_addr | global_addr[56:0] | type=2
        v4u g0;
        g0[0] = 1u;
        g0[1] = lds;
        g0[2] = (unsigned)(ga & 0xffffffffu);
        g0[3] = (unsigned)((ga >> 32) & 0x01ffffffu) | (2u << 30);
        // D# group 1: data_size=4B(2) | pad_en | pad_interval=128dw(6) |
        //             pad_amount=2dw(1) | dims/strides
        v8i g1;
        g1[0] = (int)((2u << 16) | (1u << 20) | (6u << 22) | (1u << 25));
        g1[1] = (int)(128u << 16);        // tensor_dim0 = 128
        g1[2] = (int)(64u << 16);         // tensor_dim1 = 64
        g1[3] = (int)(128u << 16);        // tile_dim0 = 128
        g1[4] = (int)64;                  // tile_dim1 = 64, tile_dim2 = 0
        g1[5] = (int)HW;                  // tensor_dim0_stride = 16384
        g1[6] = 0;
        g1[7] = 0;
        v4i gz  = {0, 0, 0, 0};           // groups 2/3 unused (2D tensor)
        v8i gz8 = {0, 0, 0, 0, 0, 0, 0, 0};
        __builtin_amdgcn_tensor_load_to_lds(g0, g1, gz, gz, gz8, 0);
        __builtin_amdgcn_s_wait_tensorcnt(0);
    }
    __syncthreads();

    // A fragments from LDS: pixel column = wv*16 + m, channel rows kb.
    const int pcol = wv * 16 + m;
    v16h a0, a1;
#pragma unroll
    for (int j = 0; j < 8; ++j) {
        int kb = (j < 4) ? (hl * 8 + 2 * j) : (16 + hl * 8 + 2 * (j - 4));
        a0[2 * j]     = (_Float16)shx[(kb)          * XROW + pcol];
        a0[2 * j + 1] = (_Float16)shx[(kb + 1)      * XROW + pcol];
        a1[2 * j]     = (_Float16)shx[(32 + kb)     * XROW + pcol];
        a1[2 * j + 1] = (_Float16)shx[(32 + kb + 1) * XROW + pcol];
    }

    const v16h* pb  = (const v16h*)pwom;
    const int ptile = blockIdx.x * 8 + wv;
#pragma unroll
    for (int nt = 0; nt < 7; ++nt) {
        const v16h bf0 = pb[(nt * 2 + 0) * 32 + lane];
        const v16h bf1 = pb[(nt * 2 + 1) * 32 + lane];
        v8f acc = {};
        acc = __builtin_amdgcn_wmma_f32_16x16x32_f16(false, a0, false, bf0,
                                                     (short)0, acc, false, false);
        acc = __builtin_amdgcn_wmma_f32_16x16x32_f16(false, a1, false, bf1,
                                                     (short)0, acc, false, false);
        const int o    = nt * 16 + m;
        const float bs = b112[o];
#pragma unroll
        for (int r = 0; r < 8; ++r) {
            int mm = r + 8 * hl;
            om[(size_t)(ptile * 16 + mm) * OMS + o] = acc[r] + bs;
        }
    }
}

// ---------------------------------------------------------------------------
// Kernel 2: modulated deformable conv per group; 5 WMMA K-chunks of 32 over
// (ci,k)=144 padded to 160. B fragments prepacked; output staged through LDS
// so global stores are b128.
// ---------------------------------------------------------------------------
struct Samp { float w00, w01, w10, w11; int i00, i01, i10, i11; };

__device__ inline Samp make_samp(const float* __restrict__ omr, int k,
                                 float scale, int hi, int wi)
{
    Samp s;
    float offy = omr[2 * k]     * scale;
    float offx = omr[2 * k + 1] * scale;
    float msk  = 1.f / (1.f + __expf(-omr[2 * K + k]));
    float py = (float)(hi - PAD + (k / KS)) + offy;
    float px = (float)(wi - PAD + (k % KS)) + offx;
    float y0f = floorf(py), x0f = floorf(px);
    float wy1 = py - y0f, wx1 = px - x0f;
    float wy0 = 1.f - wy1, wx0 = 1.f - wx1;
    int y0 = (int)y0f, x0 = (int)x0f, y1 = y0 + 1, x1 = x0 + 1;
    bool vy0 = (y0 >= 0) && (y0 < H), vy1 = (y1 >= 0) && (y1 < H);
    bool vx0 = (x0 >= 0) && (x0 < W), vx1 = (x1 >= 0) && (x1 < W);
    int cy0 = min(max(y0, 0), H - 1), cy1 = min(max(y1, 0), H - 1);
    int cx0 = min(max(x0, 0), W - 1), cx1 = min(max(x1, 0), W - 1);
    s.w00 = (vy0 && vx0) ? msk * wy0 * wx0 : 0.f;
    s.w01 = (vy0 && vx1) ? msk * wy0 * wx1 : 0.f;
    s.w10 = (vy1 && vx0) ? msk * wy1 * wx0 : 0.f;
    s.w11 = (vy1 && vx1) ? msk * wy1 * wx1 : 0.f;
    s.i00 = cy0 * W + cx0; s.i01 = cy0 * W + cx1;
    s.i10 = cy1 * W + cx0; s.i11 = cy1 * W + cx1;
    return s;
}

__device__ inline float samp_get(const Samp& s, const float* __restrict__ xg, int ci)
{
    const float* p = xg + (size_t)ci * HW;
    return s.w00 * p[s.i00] + s.w01 * p[s.i01] +
           s.w10 * p[s.i10] + s.w11 * p[s.i11];
}

__global__ __launch_bounds__(256) void dcn_wmma(
    const float* __restrict__ x, const float* __restrict__ om,
    const _Float16* __restrict__ pwdcn, const float* __restrict__ dcn_b,
    const float* __restrict__ offset_scale, float* __restrict__ outp)
{
    const int lane  = threadIdx.x & 31;
    const int wv    = threadIdx.x >> 5;
    const int ptile = blockIdx.x * 8 + wv;
    const int g     = blockIdx.y;
    const int m     = lane & 15;          // pixel row of A / co column of B,C
    const int hl    = lane >> 4;
    const float scale = offset_scale[0];

    const int pix = ptile * 16 + m;
    const int b   = pix >> 14;
    const int hw  = pix & (HW - 1);
    const int hi  = hw >> 7, wi = hw & (W - 1);

    const float* omr = om + (size_t)pix * OMS + g * (3 * K);
    const float* xg  = x + ((size_t)b * C_IN + g * CG) * HW;
    const v16h* pb   = (const v16h*)pwdcn + (size_t)g * 5 * 32;
    __builtin_prefetch(omr, 0, 0);  // global_prefetch_b8 on the offset row

    v8f acc = {};
#pragma unroll
    for (int c = 0; c < 5; ++c) {     // reduction r = k*16+ci, chunks of 32
        const int k0 = 2 * c, k1 = 2 * c + 1;
        Samp sA = make_samp(omr, k0, scale, hi, wi);
        Samp sB;
        if (k1 < K) sB = make_samp(omr, k1, scale, hi, wi);
        else { sB.w00 = sB.w01 = sB.w10 = sB.w11 = 0.f;
               sB.i00 = sB.i01 = sB.i10 = sB.i11 = 0; }

        const v16h bf = pb[c * 32 + lane];

        v16h a;
#pragma unroll
        for (int j = 0; j < 4; ++j) {
            int ci = 8 * hl + 2 * j;     // j<4 -> k0 half, j>=4 -> k1 half
            a[2 * j]           = (_Float16)samp_get(sA, xg, ci);
            a[2 * j + 1]       = (_Float16)samp_get(sA, xg, ci + 1);
            a[2 * (j + 4)]     = (_Float16)samp_get(sB, xg, ci);
            a[2 * (j + 4) + 1] = (_Float16)samp_get(sB, xg, ci + 1);
        }
        acc = __builtin_amdgcn_wmma_f32_16x16x32_f16(false, a, false, bf,
                                                     (short)0, acc, false, false);
    }

    // Stage C tile in LDS (pixel-major rows, padded), store back as b128.
    __shared__ float sh[8][16][17];
    const float bias = dcn_b[g * CG + m];
#pragma unroll
    for (int r = 0; r < 8; ++r)
        sh[wv][r + 8 * hl][m] = acc[r] + bias;
    __syncthreads();

    const int co = m;                 // channel this lane writes
    const int p0 = hl * 8;            // 8 pixels per lane
    float vbuf[8];
#pragma unroll
    for (int p = 0; p < 8; ++p)
        vbuf[p] = sh[wv][p0 + p][co];

    const int pbase = ptile * 16 + p0;
    const int b2 = pbase >> 14, hw2 = pbase & (HW - 1);
    float* dst = outp + ((size_t)b2 * C_OUT + g * CG + co) * HW + hw2;
    *(float4*)(dst)     = *(float4*)&vbuf[0];
    *(float4*)(dst + 4) = *(float4*)&vbuf[4];
}

// ---------------------------------------------------------------------------
// Kernel 3: GroupNorm statistics per (b,g): mean and rsqrt(var+eps)
// ---------------------------------------------------------------------------
__global__ __launch_bounds__(256) void gn_stats(const float* __restrict__ v,
                                                float* __restrict__ stats)
{
    const int bg = blockIdx.x;             // b*G + g
    const int b = bg >> 2, g = bg & 3;
    const float* base = v + ((size_t)b * C_OUT + g * CG) * HW;
    const int N = CG * HW;                 // 262144
    float s = 0.f, s2 = 0.f;
    for (int i = threadIdx.x; i < N; i += 256) {
        float t = base[i];
        s += t; s2 += t * t;
    }
    __shared__ float sh[256], sh2[256];
    sh[threadIdx.x] = s; sh2[threadIdx.x] = s2;
    __syncthreads();
    for (int ofs = 128; ofs > 0; ofs >>= 1) {
        if (threadIdx.x < ofs) {
            sh[threadIdx.x]  += sh[threadIdx.x + ofs];
            sh2[threadIdx.x] += sh2[threadIdx.x + ofs];
        }
        __syncthreads();
    }
    if (threadIdx.x == 0) {
        float mu  = sh[0] / (float)N;
        float var = sh2[0] / (float)N - mu * mu;
        stats[2 * bg]     = mu;
        stats[2 * bg + 1] = rsqrtf(var + EPS);
    }
}

// ---------------------------------------------------------------------------
// Kernel 4: normalize + affine + exact GELU
// ---------------------------------------------------------------------------
__global__ __launch_bounds__(256) void gn_gelu(
    const float* __restrict__ v, const float* __restrict__ stats,
    const float* __restrict__ gw, const float* __restrict__ gb,
    float* __restrict__ out)
{
    int idx = blockIdx.x * 256 + threadIdx.x;
    if (idx >= Bn * C_OUT * HW) return;
    int c  = (idx >> 14) & (C_OUT - 1);
    int b  = idx >> 20;
    int bg = b * G + (c >> 4);
    float t = (v[idx] - stats[2 * bg]) * stats[2 * bg + 1];
    t = t * gw[c] + gb[c];
    out[idx] = 0.5f * t * (1.f + erff(t * 0.70710678118654752f));
}

// ---------------------------------------------------------------------------
// Launch. Workspace (floats): om [NPIX*OMS] | conv [Bn*C_OUT*HW] | stats [32]
// | b112 [112] | pwom [7168 f16] | pwdcn [10240 f16]  -> ~46.2 MB of d_ws.
// ---------------------------------------------------------------------------
extern "C" void kernel_launch(void* const* d_in, const int* in_sizes, int n_in,
                              void* d_out, int out_size, void* d_ws, size_t ws_size,
                              hipStream_t stream)
{
    const float* x    = (const float*)d_in[0];
    const float* w_om = (const float*)d_in[1];
    const float* b_om = (const float*)d_in[2];
    const float* dw   = (const float*)d_in[3];
    const float* db   = (const float*)d_in[4];
    const float* gw   = (const float*)d_in[5];
    const float* gb   = (const float*)d_in[6];
    const float* osc  = (const float*)d_in[7];

    float* ws      = (float*)d_ws;
    float* om_ws   = ws;                                  // NPIX * OMS
    float* conv_ws = ws + (size_t)NPIX * OMS;             // Bn * C_OUT * HW
    float* stats   = conv_ws + (size_t)Bn * C_OUT * HW;   // 32
    float* b112    = stats + 32;                          // 112
    _Float16* pwom  = (_Float16*)(b112 + OMS);            // 7168 f16 (32B aligned)
    _Float16* pwdcn = pwom + PWOM_N;                      // 10240 f16

    prepack<<<1, 256, 0, stream>>>(w_om, b_om, dw, pwom, pwdcn, b112);

    om_gemm_wmma<<<NPIX / 128, 256, 0, stream>>>(x, pwom, b112, om_ws);

    dim3 g2(NPIX / 16 / 8, G);                            // (512, 4)
    dcn_wmma<<<g2, 256, 0, stream>>>(x, om_ws, pwdcn, db, osc, conv_ws);

    gn_stats<<<Bn * G, 256, 0, stream>>>(conv_ws, stats);

    gn_gelu<<<(Bn * C_OUT * HW) / 256, 256, 0, stream>>>(conv_ws, stats, gw, gb,
                                                         (float*)d_out);
}
